// ScaledDotProductAttention_41223096107668
// MI455X (gfx1250) — matrix operand order repaired and checked by reading, not compiled, not run on hardware
//
#include <hip/hip_runtime.h>

typedef __attribute__((ext_vector_type(16))) _Float16 v16h;
typedef __attribute__((ext_vector_type(8)))  float    v8f;

#define S_LEN 2048
#define D_DIM 64

static __device__ __forceinline__ v8f wmma_f16(const v16h& a, const v16h& b, const v8f& c) {
    return __builtin_amdgcn_wmma_f32_16x16x32_f16(false, a, false, b, (short)0, c, false, false);
}

// DPP16 lane shuffle, stays in the VALU pipe (no ds_bpermute round trip).
template <int CTRL>
static __device__ __forceinline__ float dppf(float x) {
    return __int_as_float(__builtin_amdgcn_update_dpp(
        0, __float_as_int(x), CTRL, 0xf, 0xf, true));
}

// Butterfly reduction over each group of 16 lanes; result broadcast to all 16.
static __device__ __forceinline__ float red_max16(float x) {
    x = fmaxf(x, dppf<0xB1>(x));     // quad_perm [1,0,3,2]  (xor 1)
    x = fmaxf(x, dppf<0x4E>(x));     // quad_perm [2,3,0,1]  (xor 2)
    x = fmaxf(x, dppf<0x141>(x));    // row_half_mirror      (xor 7 within 8)
    x = fmaxf(x, dppf<0x140>(x));    // row_mirror           (xor 15 within 16)
    return x;
}
static __device__ __forceinline__ float red_add16(float x) {
    x += dppf<0xB1>(x);
    x += dppf<0x4E>(x);
    x += dppf<0x141>(x);
    x += dppf<0x140>(x);
    return x;
}

__global__ __launch_bounds__(32)
void attn_fwd_kernel(const float* __restrict__ q,
                     const float* __restrict__ k,
                     const float* __restrict__ v,
                     float* __restrict__ out_o,
                     float* __restrict__ out_w)
{
    __shared__ float ldsP[16][36];            // 16x32 P tile, padded stride (144B rows)

    const int lane = threadIdx.x;
    const int n16  = lane & 15;               // column / N index
    const int hi   = lane >> 4;               // which 16-lane half
    const int it   = blockIdx.x;              // query tile 0..127
    const int bh   = blockIdx.y;              // fused (b,h) 0..63
    const int r0   = it * 16;

    const float* Q = q + (size_t)bh * S_LEN * D_DIM;
    const float* K = k + (size_t)bh * S_LEN * D_DIM;
    const float* V = v + (size_t)bh * S_LEN * D_DIM;
    float* O = out_o + ((size_t)bh * S_LEN + r0) * D_DIM;
    float* W = out_w + (size_t)bh * S_LEN * S_LEN;

    const float SCALE = 0.125f;               // 1/sqrt(64), folded into Q fragment
    const float MASK_NEG = -1.0e9f;

    // ---- Q A-fragments, pre-scaled by 1/sqrt(D). Loaded once. ----
    v16h qa0, qa1;
    {
        const float* qrow = Q + (size_t)(r0 + n16) * D_DIM;
        #pragma unroll
        for (int u = 0; u < 8; ++u) {
            const int kb = 2 * u + 16 * hi;                     // A-layout K pair base (K = 16*hi + 2u)
            float2 f0 = *(const float2*)(qrow + kb);
            float2 f1 = *(const float2*)(qrow + 32 + kb);
            qa0[2*u] = (_Float16)(f0.x * SCALE); qa0[2*u+1] = (_Float16)(f0.y * SCALE);
            qa1[2*u] = (_Float16)(f1.x * SCALE); qa1[2*u+1] = (_Float16)(f1.y * SCALE);
        }
    }

    const int njt   = (r0 + 15) / 32 + 1;                         // all key steps touched
    const int nfull = (r0 >= 31) ? ((r0 - 31) / 32 + 1) : 0;      // steps needing no mask

    // Two 16x16 score tiles for cols c0..c0+31 (already scaled via Q).
    auto compute_scores = [&](int c0, v8f& s0, v8f& s1) {
        #pragma unroll
        for (int t = 0; t < 2; ++t) {
            v8f acc = {};
            const float* krow = K + (size_t)(c0 + 16 * t + n16) * D_DIM;  // B col n16
            v16h kb;
            #pragma unroll
            for (int u = 0; u < 8; ++u) {      // features 0..31 (K pair = 16*hi + 2u)
                float2 f = *(const float2*)(krow + 16 * hi + 2 * u);
                kb[2*u] = (_Float16)f.x; kb[2*u+1] = (_Float16)f.y;
            }
            acc = wmma_f16(qa0, kb, acc);
            #pragma unroll
            for (int u = 0; u < 8; ++u) {      // features 32..63
                float2 f = *(const float2*)(krow + 32 + 16 * hi + 2 * u);
                kb[2*u] = (_Float16)f.x; kb[2*u+1] = (_Float16)f.y;
            }
            acc = wmma_f16(qa1, kb, acc);
            if (t == 0) s0 = acc; else s1 = acc;
        }
    };

    // Causal mask for the diagonal tile only.
    auto apply_mask = [&](int c0, v8f& s0, v8f& s1) {
        #pragma unroll
        for (int vv = 0; vv < 8; ++vv) {       // C layout: row = r0+vv+8*hi, col = c0+16t+n16
            const int row = r0 + vv + 8 * hi;
            if (c0 + n16 > row)      s0[vv] = MASK_NEG;
            if (c0 + 16 + n16 > row) s1[vv] = MASK_NEG;
        }
    };

    // ---------------- Pass 1: row max m; per-lane partial sum-exp l ----------------
    float m_s[8], l_s[8];
    #pragma unroll
    for (int vv = 0; vv < 8; ++vv) { m_s[vv] = -3.0e38f; l_s[vv] = 0.0f; }

    // m is reduced (lane-uniform per row) every step, so the rescale factor is
    // uniform across lanes and l can stay a per-lane partial sum until the end.
    auto update_stats = [&](const v8f& s0, const v8f& s1) {
        #pragma unroll
        for (int vv = 0; vv < 8; ++vv) {
            const float mt = red_max16(fmaxf(s0[vv], s1[vv]));
            const float mn = fmaxf(m_s[vv], mt);
            l_s[vv] = l_s[vv] * __expf(m_s[vv] - mn)
                    + __expf(s0[vv] - mn) + __expf(s1[vv] - mn);
            m_s[vv] = mn;
        }
    };

    for (int jt = 0; jt < nfull; ++jt) {                     // interior: no masking
        const int c0 = jt * 32;
        if (jt + 1 < njt)
            __builtin_prefetch(K + (size_t)(c0 + 32 + lane) * D_DIM, 0, 1);
        v8f s0, s1;
        compute_scores(c0, s0, s1);
        update_stats(s0, s1);
    }
    for (int jt = nfull; jt < njt; ++jt) {                   // diagonal: masked
        const int c0 = jt * 32;
        v8f s0, s1;
        compute_scores(c0, s0, s1);
        apply_mask(c0, s0, s1);
        update_stats(s0, s1);
    }

    float rl[8];
    #pragma unroll
    for (int vv = 0; vv < 8; ++vv)
        rl[vv] = 1.0f / red_add16(l_s[vv]);                  // one lane-reduce at the end

    // ---------------- Pass 2: weights out + O = P @ V ----------------
    v8f o_acc[4];
    #pragma unroll
    for (int cc = 0; cc < 4; ++cc) o_acc[cc] = (v8f){};

    auto emit_pv = [&](int c0, const v8f& s0, const v8f& s1) {
        // normalized weights: write to global W and to LDS for layout transpose
        #pragma unroll
        for (int vv = 0; vv < 8; ++vv) {
            const float w0 = __expf(s0[vv] - m_s[vv]) * rl[vv];   // masked -> exactly 0
            const float w1 = __expf(s1[vv] - m_s[vv]) * rl[vv];
            const int row = r0 + vv + 8 * hi;
            W[(size_t)row * S_LEN + c0 + n16]      = w0;
            W[(size_t)row * S_LEN + c0 + 16 + n16] = w1;
            ldsP[vv + 8 * hi][n16]      = w0;
            ldsP[vv + 8 * hi][16 + n16] = w1;
        }
        // re-read P in A-layout as f16 (same-wave DS ops are in-order)
        v16h pa;
        #pragma unroll
        for (int u = 0; u < 8; ++u) {
            const int kb = 2 * u + 16 * hi;
            pa[2*u]   = (_Float16)ldsP[n16][kb];
            pa[2*u+1] = (_Float16)ldsP[n16][kb + 1];
        }
        // O[16x64] += P[16x32] @ V[32x64], four 16-col output chunks
        #pragma unroll
        for (int cc = 0; cc < 4; ++cc) {
            v16h vb;                            // B layout: lane=n16, K = 16*hi + 2u(+1)
            #pragma unroll
            for (int u = 0; u < 8; ++u) {
                vb[2*u]   = (_Float16)V[(size_t)(c0 + 16 * hi + 2 * u)     * D_DIM + cc * 16 + n16];
                vb[2*u+1] = (_Float16)V[(size_t)(c0 + 16 * hi + 2 * u + 1) * D_DIM + cc * 16 + n16];
            }
            o_acc[cc] = wmma_f16(pa, vb, o_acc[cc]);
        }
    };

    for (int jt = 0; jt < nfull; ++jt) {                     // interior: no masking
        const int c0 = jt * 32;
        __builtin_prefetch(V + (size_t)(c0 + lane) * D_DIM, 0, 1);
        v8f s0, s1;
        compute_scores(c0, s0, s1);
        emit_pv(c0, s0, s1);
    }
    for (int jt = nfull; jt < njt; ++jt) {                   // diagonal: masked
        const int c0 = jt * 32;
        v8f s0, s1;
        compute_scores(c0, s0, s1);
        apply_mask(c0, s0, s1);
        emit_pv(c0, s0, s1);
    }

    // ---- store O ----
    #pragma unroll
    for (int cc = 0; cc < 4; ++cc)
        #pragma unroll
        for (int vv = 0; vv < 8; ++vv)
            O[(size_t)(vv + 8 * hi) * D_DIM + cc * 16 + n16] = o_acc[cc][vv];

    // ---- zero-fill fully-masked region of W (cols >= njt*32) ----
    const int cz0 = njt * 32;
    if (cz0 < S_LEN) {
        const float4 z = make_float4(0.f, 0.f, 0.f, 0.f);
        for (int r = 0; r < 16; ++r) {
            float* wr = W + (size_t)(r0 + r) * S_LEN;
            for (int c = cz0 + lane * 4; c < S_LEN; c += 128)
                *(float4*)(wr + c) = z;
        }
    }
}

extern "C" void kernel_launch(void* const* d_in, const int* in_sizes, int n_in,
                              void* d_out, int out_size, void* d_ws, size_t ws_size,
                              hipStream_t stream) {
    const float* q = (const float*)d_in[0];
    const float* k = (const float*)d_in[1];
    const float* v = (const float*)d_in[2];
    // d_in[3] is the causal mask; causality is applied analytically in-kernel.
    float* out   = (float*)d_out;
    float* out_o = out;                                   // [B,H,S,D]
    float* out_w = out + (size_t)4 * 16 * 2048 * 64;      // [B,H,S,S]

    dim3 grid(S_LEN / 16, 4 * 16);   // 128 query tiles x 64 (b,h) heads
    attn_fwd_kernel<<<grid, 32, 0, stream>>>(q, k, v, out_o, out_w);
}
